// MixtralSparseMoeBlock_85710367359159
// MI455X (gfx1250) — compile-verified, matrix operand order checked by reading, not run on hardware
//
#include <hip/hip_runtime.h>
#include <hip/hip_bf16.h>

#define T_TOK 8192
#define HDIM  1024
#define IDIM  4096
#define NEXP  8
#define KTOP  2

#define BM 128
#define BN 128
#define BK 32
#define LDP 40   // padded LDS row (elements): 80B = 5*16B, b128-aligned, conflict-free

typedef __attribute__((ext_vector_type(16))) __bf16 v16bf;
typedef __attribute__((ext_vector_type(8)))  __bf16 bf16x8;
typedef __attribute__((ext_vector_type(4)))  __bf16 bf16x4;
typedef __attribute__((ext_vector_type(8)))  float  v8f;

union FragU { v16bf v; bf16x8 h[2]; };

__device__ __forceinline__ bf16x4 cvt4(float4 f) {
  bf16x4 r;
  r.x = (__bf16)f.x; r.y = (__bf16)f.y; r.z = (__bf16)f.z; r.w = (__bf16)f.w;
  return r;
}

__device__ __forceinline__ void zero8(__bf16* p) {
  bf16x8 z;
#pragma unroll
  for (int i = 0; i < 8; ++i) z[i] = (__bf16)0.f;
  *(bf16x8*)p = z;
}

// issue two async b128 global->LDS copies (16B each), tracked by ASYNCcnt
__device__ __forceinline__ void async_cp32(unsigned lds, const __bf16* g) {
  asm volatile(
      "global_load_async_to_lds_b128 %0, %2, off\n\t"
      "global_load_async_to_lds_b128 %1, %3, off"
      :
      : "v"(lds), "v"(lds + 16u),
        "v"((unsigned long long)g), "v"((unsigned long long)(g + 8))
      : "memory");
}

// ---------------------------------------------------------------- zero
__global__ void moe_zero_kernel(float* __restrict__ out, int* __restrict__ counters) {
  const size_t n = (size_t)T_TOK * HDIM;
  const size_t stride = (size_t)gridDim.x * blockDim.x;
  for (size_t i = (size_t)blockIdx.x * blockDim.x + threadIdx.x; i < n; i += stride)
    out[i] = 0.f;
  if (blockIdx.x == 0 && threadIdx.x < 16) counters[threadIdx.x] = 0;  // count[8]+cursor[8]
}

// ---------------------------------------------------------------- one-time fp32->bf16 weight cast
__global__ void moe_castw_kernel(const float* __restrict__ src, __bf16* __restrict__ dst,
                                 unsigned n4) {
  const size_t stride = (size_t)gridDim.x * blockDim.x;
  for (size_t i = (size_t)blockIdx.x * blockDim.x + threadIdx.x; i < n4; i += stride) {
    float4 v = ((const float4*)src)[i];
    *(bf16x4*)(dst + i * 4) = cvt4(v);
  }
}

// ---------------------------------------------------------------- router
__global__ __launch_bounds__(256) void moe_router_kernel(
    const float* __restrict__ x, const float* __restrict__ gw,
    float* __restrict__ rlog, int* __restrict__ cnt,
    int* __restrict__ tok_e, float* __restrict__ tok_w) {
  __shared__ float s_gw[NEXP * HDIM];
  const int tid = threadIdx.x;
  for (int i = tid; i < NEXP * HDIM; i += 256) s_gw[i] = gw[i];
  __syncthreads();

  const int t = blockIdx.x * 256 + tid;
  const float4* x4 = (const float4*)(x + (size_t)t * HDIM);
  float acc[NEXP];
#pragma unroll
  for (int e = 0; e < NEXP; ++e) acc[e] = 0.f;
  for (int h4 = 0; h4 < HDIM / 4; ++h4) {
    float4 xv = x4[h4];
#pragma unroll
    for (int e = 0; e < NEXP; ++e) {
      const float* g = &s_gw[e * HDIM + h4 * 4];
      acc[e] += xv.x * g[0] + xv.y * g[1] + xv.z * g[2] + xv.w * g[3];
    }
  }
#pragma unroll
  for (int e = 0; e < NEXP; ++e) rlog[(size_t)t * NEXP + e] = acc[e];

  // top-2 (ties -> lower index, matching lax.top_k); softmax denom cancels in renorm
  int i1 = 0; float l1 = acc[0];
#pragma unroll
  for (int e = 1; e < NEXP; ++e) if (acc[e] > l1) { l1 = acc[e]; i1 = e; }
  int i2 = (i1 == 0) ? 1 : 0; float l2 = acc[i2];
#pragma unroll
  for (int e = 0; e < NEXP; ++e)
    if (e != i1 && e != i2 && acc[e] > l2) { l2 = acc[e]; i2 = e; }
  float p2 = __expf(l2 - l1);          // p1 == 1
  float inv = __builtin_amdgcn_rcpf(1.f + p2);
  tok_e[t * 2 + 0] = i1; tok_w[t * 2 + 0] = inv;
  tok_e[t * 2 + 1] = i2; tok_w[t * 2 + 1] = p2 * inv;
  atomicAdd(&cnt[i1], 1);
  atomicAdd(&cnt[i2], 1);
}

// ---------------------------------------------------------------- scan (E=8)
__global__ void moe_scan_kernel(const int* __restrict__ cnt, int* __restrict__ offs,
                                int* __restrict__ cur) {
  if (threadIdx.x == 0) {
    int o = 0;
    for (int e = 0; e < NEXP; ++e) { offs[e] = o; o += cnt[e]; cur[e] = 0; }
  }
}

// ---------------------------------------------------------------- fill gathered lists
__global__ __launch_bounds__(256) void moe_fill_kernel(
    const int* __restrict__ tok_e, const float* __restrict__ tok_w,
    const int* __restrict__ offs, int* __restrict__ cur,
    int* __restrict__ ptok, float* __restrict__ pw) {
  const int t = blockIdx.x * 256 + threadIdx.x;
#pragma unroll
  for (int k = 0; k < KTOP; ++k) {
    int e = tok_e[t * 2 + k];
    int slot = offs[e] + atomicAdd(&cur[e], 1);
    ptok[slot] = t;
    pw[slot]   = tok_w[t * 2 + k];
  }
}

// ---------------------------------------------------------------- pass A: h = silu(x@w1^T)*(x@w3^T)
// w1/w3 tiles: bf16 in ws -> pure GLOBAL_LOAD_ASYNC_TO_LDS_B128 DMA, double-buffered.
// x tile: gathered fp32 rows -> VALU convert path (small: 1/3 of tile bytes).
__global__ __launch_bounds__(256) void moe_mlp1_kernel(
    const float* __restrict__ x, const __bf16* __restrict__ w1b,
    const __bf16* __restrict__ w3b, const int* __restrict__ cnt,
    const int* __restrict__ eoff, const int* __restrict__ ptok,
    __bf16* __restrict__ hbuf) {
  const int e = blockIdx.z;
  const int n_e = cnt[e];
  const int mBase = blockIdx.y * BM;
  if (mBase >= n_e) return;
  const int iBase = blockIdx.x * BN;
  const int eO = eoff[e];

  __shared__ __bf16 As [2][BM][LDP];
  __shared__ __bf16 Bs1[2][BN][LDP];
  __shared__ __bf16 Bs3[2][BN][LDP];

  const int tid = threadIdx.x;
  const int lr = tid >> 3;           // x-tile loader: 0..31
  const int lc = (tid & 7) << 2;     // 0,4,...,28
  const int ar = tid >> 1;           // async loader: 0..127
  const int ac = (tid & 1) << 4;     // 0 or 16

  int tokR[4];
#pragma unroll
  for (int it = 0; it < 4; ++it) {
    int sl = mBase + lr + it * 32;
    tokR[it] = (sl < n_e) ? ptok[eO + sl] : 0;
  }

  const __bf16* w1row = w1b + (size_t)e * IDIM * HDIM + (size_t)(iBase + ar) * HDIM + ac;
  const __bf16* w3row = w3b + (size_t)e * IDIM * HDIM + (size_t)(iBase + ar) * HDIM + ac;

  const int lane = tid & 31;
  const int wv = tid >> 5;      // 8 waves
  const int wm = wv >> 2;       // 0..1 -> 64-row slab
  const int wn = wv & 3;        // 0..3 -> 32-col slab
  const int hs = lane >> 4;     // half select
  const int ln = lane & 15;

  auto load_tiles = [&](int k0, int buf) {
    // weight tiles: async DMA, no VGPR staging, no VALU
    async_cp32((unsigned)(size_t)&Bs1[buf][ar][ac], w1row + k0);
    async_cp32((unsigned)(size_t)&Bs3[buf][ar][ac], w3row + k0);
    // x tile: fp32 -> bf16
#pragma unroll
    for (int it = 0; it < 4; ++it) {
      const int r = lr + it * 32;
      float4 av = *(const float4*)(x + (size_t)tokR[it] * HDIM + k0 + lc);
      *(bf16x4*)&As[buf][r][lc] = cvt4(av);
    }
  };

  v8f acc1[4][2], acc3[4][2];
#pragma unroll
  for (int i = 0; i < 4; ++i)
#pragma unroll
    for (int j = 0; j < 2; ++j)
#pragma unroll
      for (int r = 0; r < 8; ++r) { acc1[i][j][r] = 0.f; acc3[i][j][r] = 0.f; }

  load_tiles(0, 0);
  asm volatile("s_wait_asynccnt 0x0" ::: "memory");
  __syncthreads();

  for (int k0 = 0; k0 < HDIM; k0 += BK) {
    const int cur = (k0 >> 5) & 1;
    if (k0 + BK < HDIM) {
      // warm WGP$ for the next x K-tile (global_prefetch_b8)
      __builtin_prefetch(x + (size_t)tokR[0] * HDIM + (k0 + BK) + lc, 0, 3);
      load_tiles(k0 + BK, cur ^ 1);
    }

    v16bf afr[4], b1fr[2], b3fr[2];
#pragma unroll
    for (int mt = 0; mt < 4; ++mt) {
      const int row = wm * 64 + mt * 16 + ln;
      FragU u;
      u.h[0] = *(const bf16x8*)&As[cur][row][hs * 8];
      u.h[1] = *(const bf16x8*)&As[cur][row][16 + hs * 8];
      afr[mt] = u.v;
    }
#pragma unroll
    for (int nt = 0; nt < 2; ++nt) {
      const int row = wn * 32 + nt * 16 + ln;
      FragU u1, u3;
      u1.h[0] = *(const bf16x8*)&Bs1[cur][row][hs * 8];
      u1.h[1] = *(const bf16x8*)&Bs1[cur][row][16 + hs * 8];
      b1fr[nt] = u1.v;
      u3.h[0] = *(const bf16x8*)&Bs3[cur][row][hs * 8];
      u3.h[1] = *(const bf16x8*)&Bs3[cur][row][16 + hs * 8];
      b3fr[nt] = u3.v;
    }
#pragma unroll
    for (int mt = 0; mt < 4; ++mt)
#pragma unroll
      for (int nt = 0; nt < 2; ++nt) {
        acc1[mt][nt] = __builtin_amdgcn_wmma_f32_16x16x32_bf16(
            false, afr[mt], false, b1fr[nt], (short)0, acc1[mt][nt], false, false);
        acc3[mt][nt] = __builtin_amdgcn_wmma_f32_16x16x32_bf16(
            false, afr[mt], false, b3fr[nt], (short)0, acc3[mt][nt], false, false);
      }

    asm volatile("s_wait_asynccnt 0x0" ::: "memory");
    __syncthreads();
  }

#pragma unroll
  for (int mt = 0; mt < 4; ++mt)
#pragma unroll
    for (int nt = 0; nt < 2; ++nt)
#pragma unroll
      for (int r = 0; r < 8; ++r) {
        const int sl = mBase + wm * 64 + mt * 16 + r + 8 * hs;
        if (sl < n_e) {
          float g  = acc1[mt][nt][r];
          float uu = acc3[mt][nt][r];
          // silu(g)*u with v_rcp_f32 instead of IEEE divide (bf16 output anyway)
          float hv = g * __builtin_amdgcn_rcpf(1.f + __expf(-g)) * uu;
          const int col = iBase + wn * 32 + nt * 16 + ln;
          hbuf[(size_t)(eO + sl) * IDIM + col] = (__bf16)hv;
        }
      }
}

// ---------------------------------------------------------------- pass B: out += w * (h @ w2^T)
// Both tiles are bf16 in global -> both go through async DMA; the inner loop has
// no conversion VALU at all: async issue -> ds_load_b128 frags -> WMMA.
__global__ __launch_bounds__(256) void moe_mlp2_kernel(
    const __bf16* __restrict__ hbuf, const __bf16* __restrict__ w2b,
    const int* __restrict__ cnt, const int* __restrict__ eoff,
    const int* __restrict__ ptok, const float* __restrict__ pw,
    float* __restrict__ out) {
  const int e = blockIdx.z;
  const int n_e = cnt[e];
  const int mBase = blockIdx.y * BM;
  if (mBase >= n_e) return;
  const int nBase = blockIdx.x * BN;   // over HDIM
  const int eO = eoff[e];

  __shared__ __bf16 As[2][BM][LDP];
  __shared__ __bf16 Bs[2][BN][LDP];

  const int tid = threadIdx.x;
  const int ar = tid >> 1;           // async loader: 0..127
  const int ac = (tid & 1) << 4;     // 0 or 16
  const int lane = tid & 31;
  const int wv = tid >> 5;
  const int wm = wv >> 2;
  const int wn = wv & 3;
  const int hs = lane >> 4;
  const int ln = lane & 15;

  const int aValid = (mBase + ar) < n_e;
  const __bf16* aRow = hbuf + (size_t)(eO + mBase + ar) * IDIM + ac;
  const __bf16* bRow = w2b + (size_t)e * HDIM * IDIM + (size_t)(nBase + ar) * IDIM + ac;

  auto load_tiles = [&](int k0, int buf) {
    if (aValid) {
      async_cp32((unsigned)(size_t)&As[buf][ar][ac], aRow + k0);
    } else {
      zero8(&As[buf][ar][ac]);
      zero8(&As[buf][ar][ac + 8]);
    }
    async_cp32((unsigned)(size_t)&Bs[buf][ar][ac], bRow + k0);
  };

  v8f acc[4][2];
#pragma unroll
  for (int i = 0; i < 4; ++i)
#pragma unroll
    for (int j = 0; j < 2; ++j)
#pragma unroll
      for (int r = 0; r < 8; ++r) acc[i][j][r] = 0.f;

  load_tiles(0, 0);
  asm volatile("s_wait_asynccnt 0x0" ::: "memory");
  __syncthreads();

  for (int k0 = 0; k0 < IDIM; k0 += BK) {
    const int cur = (k0 >> 5) & 1;
    if (k0 + BK < IDIM) load_tiles(k0 + BK, cur ^ 1);

    v16bf afr[4], bfr[2];
#pragma unroll
    for (int mt = 0; mt < 4; ++mt) {
      const int row = wm * 64 + mt * 16 + ln;
      FragU u;
      u.h[0] = *(const bf16x8*)&As[cur][row][hs * 8];
      u.h[1] = *(const bf16x8*)&As[cur][row][16 + hs * 8];
      afr[mt] = u.v;
    }
#pragma unroll
    for (int nt = 0; nt < 2; ++nt) {
      const int row = wn * 32 + nt * 16 + ln;
      FragU u;
      u.h[0] = *(const bf16x8*)&Bs[cur][row][hs * 8];
      u.h[1] = *(const bf16x8*)&Bs[cur][row][16 + hs * 8];
      bfr[nt] = u.v;
    }
#pragma unroll
    for (int mt = 0; mt < 4; ++mt)
#pragma unroll
      for (int nt = 0; nt < 2; ++nt)
        acc[mt][nt] = __builtin_amdgcn_wmma_f32_16x16x32_bf16(
            false, afr[mt], false, bfr[nt], (short)0, acc[mt][nt], false, false);

    // next buffer's async copies must land in LDS before anyone reads it
    asm volatile("s_wait_asynccnt 0x0" ::: "memory");
    __syncthreads();
  }

#pragma unroll
  for (int mt = 0; mt < 4; ++mt)
#pragma unroll
    for (int r = 0; r < 8; ++r) {
      const int sl = mBase + wm * 64 + mt * 16 + r + 8 * hs;
      if (sl < n_e) {
        const int gs = eO + sl;
        const int tok = ptok[gs];
        const float wgt = pw[gs];
#pragma unroll
        for (int nt = 0; nt < 2; ++nt) {
          const int col = nBase + wn * 32 + nt * 16 + ln;
          unsafeAtomicAdd(&out[(size_t)tok * HDIM + col], acc[mt][nt][r] * wgt);
        }
      }
    }
}

// ---------------------------------------------------------------- launch
extern "C" void kernel_launch(void* const* d_in, const int* in_sizes, int n_in,
                              void* d_out, int out_size, void* d_ws, size_t ws_size,
                              hipStream_t stream) {
  const float* x  = (const float*)d_in[0];
  const float* gw = (const float*)d_in[1];
  const float* w1 = (const float*)d_in[2];
  const float* w2 = (const float*)d_in[3];
  const float* w3 = (const float*)d_in[4];
  float* out  = (float*)d_out;
  float* rlog = out + (size_t)T_TOK * HDIM;

  const size_t WELT = (size_t)NEXP * IDIM * HDIM;   // elements per weight tensor

  // workspace layout
  int* count   = (int*)d_ws;                       // [8]
  int* cursor  = count + 8;                        // [8]
  int* offs    = count + 16;                       // [8]
  int* tok_e   = count + 32;                       // [T*2]
  float* tok_w = (float*)(tok_e + T_TOK * KTOP);   // [T*2]
  int* ptok    = (int*)(tok_w + T_TOK * KTOP);     // [T*2]
  float* pw    = (float*)(ptok + T_TOK * KTOP);    // [T*2]
  __bf16* hbuf = (__bf16*)(pw + T_TOK * KTOP);     // [T*2][IDIM] bf16
  __bf16* w1b  = hbuf + (size_t)T_TOK * KTOP * IDIM;
  __bf16* w3b  = w1b + WELT;
  __bf16* w2b  = w3b + WELT;

  moe_zero_kernel<<<2048, 256, 0, stream>>>(out, count);
  moe_castw_kernel<<<2048, 256, 0, stream>>>(w1, w1b, (unsigned)(WELT / 4));
  moe_castw_kernel<<<2048, 256, 0, stream>>>(w3, w3b, (unsigned)(WELT / 4));
  moe_castw_kernel<<<2048, 256, 0, stream>>>(w2, w2b, (unsigned)(WELT / 4));
  moe_router_kernel<<<T_TOK / 256, 256, 0, stream>>>(x, gw, rlog, count, tok_e, tok_w);
  moe_scan_kernel<<<1, 32, 0, stream>>>(count, offs, cursor);
  moe_fill_kernel<<<T_TOK / 256, 256, 0, stream>>>(tok_e, tok_w, offs, cursor, ptok, pw);
  moe_mlp1_kernel<<<dim3(IDIM / BN, T_TOK / BM, NEXP), 256, 0, stream>>>(
      x, w1b, w3b, count, offs, ptok, hbuf);
  moe_mlp2_kernel<<<dim3(HDIM / BN, T_TOK / BM, NEXP), 256, 0, stream>>>(
      hbuf, w2b, count, offs, ptok, pw, out);
}
